// LegacyRoleSensitiveEmbedding_28621662060562
// MI455X (gfx1250) — matrix operand order repaired and checked by reading, not compiled, not run on hardware
//
#include <hip/hip_runtime.h>

// ---------------------------------------------------------------------------
// Role-sensitive embedding: out = role ? (emb[id] @ R^T) : emb[id]
//
// Strategy (MI455X):
//   Pass 1a: split R (f32) into bf16 hi/lo pair once           (4 MB ws)
//   Pass 1b: gather x = emb[ids] and split into bf16 hi/lo     (64 MB ws)
//   Pass 2:  bf16x3 split GEMM on v_wmma_f32_16x16x32_bf16 with f32 accum
//            -> inner loop is just b128 loads + WMMAs (no convert VALU)
// Fallback: if ws too small, single-kernel inline-split version.
// ---------------------------------------------------------------------------

typedef __attribute__((ext_vector_type(16))) __bf16          v16bf;
typedef __attribute__((ext_vector_type(16))) unsigned short  v16u;
typedef __attribute__((ext_vector_type(8)))  unsigned short  v8u;
typedef __attribute__((ext_vector_type(8)))  float           v8f;

#define D_MODEL 1024
#define NTOK    16384      // B*L = 4*4096
#define TILE_M  64
#define TILE_N  256

// round-to-nearest-even f32 -> bf16 bit pattern
__device__ __forceinline__ unsigned short f2bf_rne(float f) {
    unsigned u = __builtin_bit_cast(unsigned, f);
    unsigned r = u + 0x7FFFu + ((u >> 16) & 1u);
    return (unsigned short)(r >> 16);
}

__device__ __forceinline__ void split1(float f, unsigned short& h, unsigned short& l) {
    h = f2bf_rne(f);
    float fh = __builtin_bit_cast(float, ((unsigned)h) << 16);
    l = f2bf_rne(f - fh);
}

__device__ __forceinline__ float4 ldg4(const float* p) {
    return *reinterpret_cast<const float4*>(p);
}

// ---------------------------------------------------------------------------
// Pass 1a: R -> Rhi/Rlo   (one float4 per thread)
// ---------------------------------------------------------------------------
__global__ __launch_bounds__(256) void split_R_kernel(
    const float* __restrict__ R,
    unsigned short* __restrict__ Rhi, unsigned short* __restrict__ Rlo)
{
    const long base = ((long)blockIdx.x * 256 + threadIdx.x) * 4;
    float4 v = ldg4(R + base);
    unsigned short h0, h1, h2, h3, l0, l1, l2, l3;
    split1(v.x, h0, l0); split1(v.y, h1, l1);
    split1(v.z, h2, l2); split1(v.w, h3, l3);
    uint2 hv, lv;
    hv.x = (unsigned)h0 | ((unsigned)h1 << 16);
    hv.y = (unsigned)h2 | ((unsigned)h3 << 16);
    lv.x = (unsigned)l0 | ((unsigned)l1 << 16);
    lv.y = (unsigned)l2 | ((unsigned)l3 << 16);
    *reinterpret_cast<uint2*>(Rhi + base) = hv;
    *reinterpret_cast<uint2*>(Rlo + base) = lv;
}

// ---------------------------------------------------------------------------
// Pass 1b: gather + split  x[tok] = emb[ids[tok]]  -> Xhi/Xlo (row-major)
// ---------------------------------------------------------------------------
__global__ __launch_bounds__(256) void split_X_kernel(
    const int* __restrict__ ids, const float* __restrict__ emb,
    unsigned short* __restrict__ Xhi, unsigned short* __restrict__ Xlo)
{
    const long idx  = (long)blockIdx.x * 256 + threadIdx.x;   // float4 index
    const long row  = idx >> (10 - 2);                        // / (D_MODEL/4)
    const long c4   = idx & ((D_MODEL / 4) - 1);
    const long id   = ids[row];
    float4 v = ldg4(emb + id * D_MODEL + c4 * 4);
    unsigned short h0, h1, h2, h3, l0, l1, l2, l3;
    split1(v.x, h0, l0); split1(v.y, h1, l1);
    split1(v.z, h2, l2); split1(v.w, h3, l3);
    uint2 hv, lv;
    hv.x = (unsigned)h0 | ((unsigned)h1 << 16);
    hv.y = (unsigned)h2 | ((unsigned)h3 << 16);
    lv.x = (unsigned)l0 | ((unsigned)l1 << 16);
    lv.y = (unsigned)l2 | ((unsigned)l3 << 16);
    const long base = row * D_MODEL + c4 * 4;
    *reinterpret_cast<uint2*>(Xhi + base) = hv;
    *reinterpret_cast<uint2*>(Xlo + base) = lv;
}

// ---------------------------------------------------------------------------
// Pass 2: main WMMA GEMM (pre-split bf16 operands)
// ---------------------------------------------------------------------------
__device__ __forceinline__ v16bf load_afrag(const unsigned short* p, int k0) {
    // A 16x32 bf16 frag: two 8-element chunks (K+0..7 and K+16..23 lane-local)
    v8u c0 = *reinterpret_cast<const v8u*>(p + k0);
    v8u c1 = *reinterpret_cast<const v8u*>(p + k0 + 16);
    v16u u = __builtin_shufflevector(c0, c1, 0, 1, 2, 3, 4, 5, 6, 7,
                                             8, 9, 10, 11, 12, 13, 14, 15);
    return __builtin_bit_cast(v16bf, u);
}

__device__ __forceinline__ v16bf load_bfrag(const unsigned short* p, int k0) {
    // B 32x16 bf16 frag: two 8-element chunks (K+0..7 and K+16..23 lane-local)
    v8u c0 = *reinterpret_cast<const v8u*>(p + k0);
    v8u c1 = *reinterpret_cast<const v8u*>(p + k0 + 16);
    v16u u = __builtin_shufflevector(c0, c1, 0, 1, 2, 3, 4, 5, 6, 7,
                                             8, 9, 10, 11, 12, 13, 14, 15);
    return __builtin_bit_cast(v16bf, u);
}

__global__ __launch_bounds__(256) void role_emb_wmma_main(
    const int*  __restrict__ ids, const int* __restrict__ roles,
    const float* __restrict__ emb,
    const unsigned short* __restrict__ Xhi, const unsigned short* __restrict__ Xlo,
    const unsigned short* __restrict__ Rhi, const unsigned short* __restrict__ Rlo,
    float* __restrict__ out)
{
    __shared__ int s_id[TILE_M];
    __shared__ int s_role[TILE_M];

    const int tid     = threadIdx.x;
    const int tokBase = blockIdx.x * TILE_M;
    if (tid < TILE_M) {
        s_id[tid]   = ids[tokBase + tid];
        s_role[tid] = roles[tokBase + tid];
    }
    __syncthreads();

    const int lane = tid & 31;
    const int wave = tid >> 5;       // 0..7
    const int wm   = wave >> 2;      // 0..1
    const int wn   = wave & 3;       // 0..3
    const int half = lane >> 4;
    const int l16  = lane & 15;

    const int nBase = blockIdx.y * TILE_N + wn * 64;

    // A fragment base pointers: rows are global token rows (already gathered)
    const long rowA0 = tokBase + wm * 32 + l16;
    const long rowA1 = rowA0 + 16;
    const unsigned short* aHi0 = Xhi + rowA0 * D_MODEL + (half << 3);
    const unsigned short* aLo0 = Xlo + rowA0 * D_MODEL + (half << 3);
    const unsigned short* aHi1 = Xhi + rowA1 * D_MODEL + (half << 3);
    const unsigned short* aLo1 = Xlo + rowA1 * D_MODEL + (half << 3);

    // B fragment base pointers: B[k,n] = R[n,k], contiguous along k
    const unsigned short* bHi[4];
    const unsigned short* bLo[4];
#pragma unroll
    for (int n = 0; n < 4; ++n) {
        const long roff = (long)(nBase + n * 16 + l16) * D_MODEL + (half << 3);
        bHi[n] = Rhi + roff;
        bLo[n] = Rlo + roff;
    }

    v8f acc[2][4];
#pragma unroll
    for (int m = 0; m < 2; ++m)
#pragma unroll
        for (int n = 0; n < 4; ++n)
            acc[m][n] = (v8f){0.f, 0.f, 0.f, 0.f, 0.f, 0.f, 0.f, 0.f};

#pragma unroll 2
    for (int k0 = 0; k0 < D_MODEL; k0 += 32) {
        if (k0 + 32 < D_MODEL) {   // prefetch next K slice (global_prefetch_b8)
            __builtin_prefetch(aHi0 + k0 + 32, 0, 1);
            __builtin_prefetch(aHi1 + k0 + 32, 0, 1);
            __builtin_prefetch(bHi[0] + k0 + 32, 0, 1);
            __builtin_prefetch(bHi[2] + k0 + 32, 0, 1);
        }

        v16bf ah[2], al[2];
        ah[0] = load_afrag(aHi0, k0);
        al[0] = load_afrag(aLo0, k0);
        ah[1] = load_afrag(aHi1, k0);
        al[1] = load_afrag(aLo1, k0);

#pragma unroll
        for (int n = 0; n < 4; ++n) {
            v16bf bh = load_bfrag(bHi[n], k0);
            v16bf bl = load_bfrag(bLo[n], k0);
#pragma unroll
            for (int m = 0; m < 2; ++m) {
                acc[m][n] = __builtin_amdgcn_wmma_f32_16x16x32_bf16(
                    false, ah[m], false, bh, (short)0, acc[m][n], false, false);
                acc[m][n] = __builtin_amdgcn_wmma_f32_16x16x32_bf16(
                    false, al[m], false, bh, (short)0, acc[m][n], false, false);
                acc[m][n] = __builtin_amdgcn_wmma_f32_16x16x32_bf16(
                    false, ah[m], false, bl, (short)0, acc[m][n], false, false);
            }
        }
    }

    // epilogue: role select
#pragma unroll
    for (int m = 0; m < 2; ++m) {
#pragma unroll
        for (int r = 0; r < 8; ++r) {
            const int mLocal = wm * 32 + m * 16 + (half ? (r + 8) : r);
            const int rowIdx = tokBase + mLocal;
            const int role   = s_role[mLocal];
            const long id    = s_id[mLocal];
#pragma unroll
            for (int n = 0; n < 4; ++n) {
                const int col = nBase + n * 16 + l16;
                float v;
                if (role == 1) v = acc[m][n][r];
                else           v = emb[id * D_MODEL + col];
                out[(long)rowIdx * D_MODEL + col] = v;
            }
        }
    }
}

// ---------------------------------------------------------------------------
// Fallback: single-kernel inline-split version (when ws is too small)
// ---------------------------------------------------------------------------
__device__ __forceinline__ void split16v(const float* __restrict__ f,
                                         v16u& hi, v16u& lo) {
#pragma unroll
    for (int i = 0; i < 16; ++i) {
        unsigned short h, l;
        split1(f[i], h, l);
        hi[i] = h; lo[i] = l;
    }
}

__global__ __launch_bounds__(256) void role_emb_wmma_inline(
    const int*  __restrict__ ids, const int* __restrict__ roles,
    const float* __restrict__ emb, const float* __restrict__ R,
    float* __restrict__ out)
{
    __shared__ int s_id[TILE_M];
    __shared__ int s_role[TILE_M];

    const int tid     = threadIdx.x;
    const int tokBase = blockIdx.x * TILE_M;
    if (tid < TILE_M) {
        s_id[tid]   = ids[tokBase + tid];
        s_role[tid] = roles[tokBase + tid];
    }
    __syncthreads();

    const int lane = tid & 31;
    const int wave = tid >> 5;
    const int wm   = wave >> 2;
    const int wn   = wave & 3;
    const int half = lane >> 4;
    const int l16  = lane & 15;
    const int nBase = blockIdx.y * TILE_N + wn * 64;

    const long idA0 = s_id[wm * 32 + l16];
    const long idA1 = s_id[wm * 32 + 16 + l16];
    const float* aPtr0 = emb + idA0 * D_MODEL + (half << 3);
    const float* aPtr1 = emb + idA1 * D_MODEL + (half << 3);
    const float* bPtr[4];
#pragma unroll
    for (int n = 0; n < 4; ++n)
        bPtr[n] = R + (long)(nBase + n * 16 + l16) * D_MODEL + (half << 3);

    v8f acc[2][4];
#pragma unroll
    for (int m = 0; m < 2; ++m)
#pragma unroll
        for (int n = 0; n < 4; ++n)
            acc[m][n] = (v8f){0.f, 0.f, 0.f, 0.f, 0.f, 0.f, 0.f, 0.f};

    for (int k0 = 0; k0 < D_MODEL; k0 += 32) {
        float af[16];
        v16u ahu[2], alu[2];
        *(float4*)&af[0]  = ldg4(aPtr0 + k0);
        *(float4*)&af[4]  = ldg4(aPtr0 + k0 + 4);
        *(float4*)&af[8]  = ldg4(aPtr0 + k0 + 16);
        *(float4*)&af[12] = ldg4(aPtr0 + k0 + 20);
        split16v(af, ahu[0], alu[0]);
        *(float4*)&af[0]  = ldg4(aPtr1 + k0);
        *(float4*)&af[4]  = ldg4(aPtr1 + k0 + 4);
        *(float4*)&af[8]  = ldg4(aPtr1 + k0 + 16);
        *(float4*)&af[12] = ldg4(aPtr1 + k0 + 20);
        split16v(af, ahu[1], alu[1]);

        v16bf ah[2], al[2];
#pragma unroll
        for (int m = 0; m < 2; ++m) {
            ah[m] = __builtin_bit_cast(v16bf, ahu[m]);
            al[m] = __builtin_bit_cast(v16bf, alu[m]);
        }

#pragma unroll
        for (int n = 0; n < 4; ++n) {
            float bfv[16];
            *(float4*)&bfv[0]  = ldg4(bPtr[n] + k0);
            *(float4*)&bfv[4]  = ldg4(bPtr[n] + k0 + 4);
            *(float4*)&bfv[8]  = ldg4(bPtr[n] + k0 + 16);
            *(float4*)&bfv[12] = ldg4(bPtr[n] + k0 + 20);
            v16u bhu, blu;
            split16v(bfv, bhu, blu);
            v16bf bh = __builtin_bit_cast(v16bf, bhu);
            v16bf bl = __builtin_bit_cast(v16bf, blu);
#pragma unroll
            for (int m = 0; m < 2; ++m) {
                acc[m][n] = __builtin_amdgcn_wmma_f32_16x16x32_bf16(
                    false, ah[m], false, bh, (short)0, acc[m][n], false, false);
                acc[m][n] = __builtin_amdgcn_wmma_f32_16x16x32_bf16(
                    false, al[m], false, bh, (short)0, acc[m][n], false, false);
                acc[m][n] = __builtin_amdgcn_wmma_f32_16x16x32_bf16(
                    false, ah[m], false, bl, (short)0, acc[m][n], false, false);
            }
        }
    }

#pragma unroll
    for (int m = 0; m < 2; ++m) {
#pragma unroll
        for (int r = 0; r < 8; ++r) {
            const int mLocal = wm * 32 + m * 16 + (half ? (r + 8) : r);
            const int rowIdx = tokBase + mLocal;
            const int role   = s_role[mLocal];
            const long id    = s_id[mLocal];
#pragma unroll
            for (int n = 0; n < 4; ++n) {
                const int col = nBase + n * 16 + l16;
                float v;
                if (role == 1) v = acc[m][n][r];
                else           v = emb[id * D_MODEL + col];
                out[(long)rowIdx * D_MODEL + col] = v;
            }
        }
    }
}

// ---------------------------------------------------------------------------
extern "C" void kernel_launch(void* const* d_in, const int* in_sizes, int n_in,
                              void* d_out, int out_size, void* d_ws, size_t ws_size,
                              hipStream_t stream) {
    const int*   ids   = (const int*)d_in[0];
    const int*   roles = (const int*)d_in[1];
    const float* emb   = (const float*)d_in[2];
    const float* R     = (const float*)d_in[3];
    float* out = (float*)d_out;

    const size_t nR = (size_t)D_MODEL * D_MODEL;   // elements
    const size_t nX = (size_t)NTOK * D_MODEL;
    const size_t need = (2 * nR + 2 * nX) * sizeof(unsigned short);

    dim3 grid(NTOK / TILE_M, D_MODEL / TILE_N);    // 256 x 4
    dim3 block(256);

    if (ws_size >= need) {
        unsigned short* Rhi = (unsigned short*)d_ws;
        unsigned short* Rlo = Rhi + nR;
        unsigned short* Xhi = Rlo + nR;
        unsigned short* Xlo = Xhi + nX;

        split_R_kernel<<<dim3(nR / (256 * 4)), block, 0, stream>>>(R, Rhi, Rlo);
        split_X_kernel<<<dim3(nX / (256 * 4)), block, 0, stream>>>(ids, emb, Xhi, Xlo);
        role_emb_wmma_main<<<grid, block, 0, stream>>>(
            ids, roles, emb, Xhi, Xlo, Rhi, Rlo, out);
    } else {
        role_emb_wmma_inline<<<grid, block, 0, stream>>>(ids, roles, emb, R, out);
    }
}
